// Model_37598143709626
// MI455X (gfx1250) — matrix operand order repaired and checked by reading, not compiled, not run on hardware
//
#include <hip/hip_runtime.h>

// ---------------------------------------------------------------------------
// MI455X / gfx1250 implementation.
//  - All dense GEMMs: v_wmma_f32_16x16x32_bf16 (bf16 inputs, fp32 accum).
//  - Edge MLP strength-reduced: rel_lin = rel_emb @ gine_ew + eb (32x256),
//    gathered per edge instead of a 300k-row GEMM.
//  - Attention: flash-style online softmax, WMMA for QK^T and PV.
//  - out_proj weight cache filled via global_load_async_to_lds_b128.
// ---------------------------------------------------------------------------

typedef __bf16 bf16;
typedef __attribute__((ext_vector_type(16))) __bf16 v16bf;
typedef __attribute__((ext_vector_type(8)))  float  v8f;

static constexpr int Nn   = 50000;   // nodes
static constexpr int INF_ = 128;     // input feats
static constexpr int Hh   = 256;     // hidden
static constexpr int Ll   = 3;       // layers
static constexpr int Ee   = 300000;  // edges
static constexpr int Rr   = 32;      // relation types
static constexpr int NCc  = 10240;   // chem nodes
static constexpr int BCH  = 512;     // chunk size
static constexpr int HEADS = 8;
static constexpr int HD   = 32;      // head dim
static constexpr int NCHUNK = NCc / BCH;   // 20

__device__ __forceinline__ bf16 tobf(float f) { return (bf16)f; }

// A-fragment (16x32, 16-bit) element k-index for vector element i, lane-half lh.
__device__ __forceinline__ int a_kidx(int i, int lh) {
  int v = i >> 1, h = i & 1;
  return ((v >> 2) << 4) + (lh << 3) + ((v & 3) << 1) + h;
}
// B-fragment (32x16, 16-bit) element k-index.
__device__ __forceinline__ int b_kidx(int i, int lh) {
  int v = i >> 1, h = i & 1;
  return ((v >> 2) << 4) + (lh << 3) + ((v & 3) << 1) + h;
}

// Load a 16-element A-fragment from a row pointer already offset by lhalf*8:
// covers k = {0..7} and {16..23} relative to that pointer -> 4 x b128.
__device__ __forceinline__ void load_a_frag(const float* __restrict__ p, v16bf& a) {
  float4 q0 = *(const float4*)(p + 0);
  float4 q1 = *(const float4*)(p + 4);
  float4 q2 = *(const float4*)(p + 16);
  float4 q3 = *(const float4*)(p + 20);
  a[0]=tobf(q0.x); a[1]=tobf(q0.y); a[2]=tobf(q0.z); a[3]=tobf(q0.w);
  a[4]=tobf(q1.x); a[5]=tobf(q1.y); a[6]=tobf(q1.z); a[7]=tobf(q1.w);
  a[8]=tobf(q2.x); a[9]=tobf(q2.y); a[10]=tobf(q2.z); a[11]=tobf(q2.w);
  a[12]=tobf(q3.x); a[13]=tobf(q3.y); a[14]=tobf(q3.z); a[15]=tobf(q3.w);
}
__device__ __forceinline__ void load_a_frag_sum(const float* __restrict__ p,
                                                const float* __restrict__ p2, v16bf& a) {
  float4 q0 = *(const float4*)(p + 0),  r0 = *(const float4*)(p2 + 0);
  float4 q1 = *(const float4*)(p + 4),  r1 = *(const float4*)(p2 + 4);
  float4 q2 = *(const float4*)(p + 16), r2 = *(const float4*)(p2 + 16);
  float4 q3 = *(const float4*)(p + 20), r3 = *(const float4*)(p2 + 20);
  a[0]=tobf(q0.x+r0.x); a[1]=tobf(q0.y+r0.y); a[2]=tobf(q0.z+r0.z); a[3]=tobf(q0.w+r0.w);
  a[4]=tobf(q1.x+r1.x); a[5]=tobf(q1.y+r1.y); a[6]=tobf(q1.z+r1.z); a[7]=tobf(q1.w+r1.w);
  a[8]=tobf(q2.x+r2.x); a[9]=tobf(q2.y+r2.y); a[10]=tobf(q2.z+r2.z); a[11]=tobf(q2.w+r2.w);
  a[12]=tobf(q3.x+r3.x); a[13]=tobf(q3.y+r3.y); a[14]=tobf(q3.z+r3.z); a[15]=tobf(q3.w+r3.w);
}
// Load a 16-element B-fragment from a pointer already offset by lhalf*8:
// covers k = {0..7} and {16..23} relative to that pointer -> 4 x b128.
__device__ __forceinline__ void load_b_frag_contig(const float* __restrict__ p, v16bf& b) {
  float4 q0 = *(const float4*)(p + 0);
  float4 q1 = *(const float4*)(p + 4);
  float4 q2 = *(const float4*)(p + 16);
  float4 q3 = *(const float4*)(p + 20);
  b[0]=tobf(q0.x); b[1]=tobf(q0.y); b[2]=tobf(q0.z); b[3]=tobf(q0.w);
  b[4]=tobf(q1.x); b[5]=tobf(q1.y); b[6]=tobf(q1.z); b[7]=tobf(q1.w);
  b[8]=tobf(q2.x); b[9]=tobf(q2.y); b[10]=tobf(q2.z); b[11]=tobf(q2.w);
  b[12]=tobf(q3.x); b[13]=tobf(q3.y); b[14]=tobf(q3.z); b[15]=tobf(q3.w);
}

// ---------------------------------------------------------------------------
// Generic WMMA GEMM: C[M,N] = act(A[M,K] (+A2) @ B[K,N] + bias) (+Resid)
// Block = 256 thr = 8 waves; block tile 64x64; wave tile 16x32 (2 WMMA acc).
// Requires: M % 16 == 0, K % 32 == 0, N % 64 == 0.
// ---------------------------------------------------------------------------
template <bool HASA2>
__global__ __launch_bounds__(256) void gemm_wmma_kernel(
    const float* __restrict__ A, const float* __restrict__ A2,
    const float* __restrict__ Bm, const float* __restrict__ bias,
    const float* __restrict__ Resid, float* __restrict__ C,
    int M, int K, int N, int relu)
{
  const int lane  = threadIdx.x & 31;
  const int wave  = threadIdx.x >> 5;
  const int lrow  = lane & 15;
  const int lhalf = lane >> 4;
  const int wr  = wave & 3;       // row-tile within block (0..3)
  const int wcg = wave >> 2;      // col 32-group (0..1)
  const int row0 = blockIdx.y * 64 + wr * 16;
  const int col0 = blockIdx.x * 64 + wcg * 32;
  if (row0 >= M) return;          // wave-uniform (M % 16 == 0)

  v8f c0 = {}, c1 = {};
  // A row pointer with the lane-half k-offset folded in.
  const float* arow  = A + (size_t)(row0 + lrow) * K + (lhalf << 3);
  const float* arow2 = HASA2 ? A2 + (size_t)(row0 + lrow) * K + (lhalf << 3) : nullptr;
  // B per-lane column base with lane-half row-offset folded in.
  const float* bbase = Bm + (size_t)(lhalf << 3) * N + col0 + lrow;

  for (int k0 = 0; k0 < K; k0 += 32) {
    if (k0 + 32 < K) __builtin_prefetch(arow + k0 + 32, 0, 3);
    v16bf a, b0, b1;
    if (HASA2) load_a_frag_sum(arow + k0, arow2 + k0, a);
    else       load_a_frag(arow + k0, a);
    const float* bp = bbase + (size_t)k0 * N;
#pragma unroll
    for (int j = 0; j < 8; ++j) {      // k = 2j + 8*(j>=4), +1 within this half
      const float* p = bp + (size_t)((j << 1) + ((j >> 2) << 3)) * N;
      b0[2*j]   = tobf(p[0]);  b0[2*j+1] = tobf(p[N]);
      b1[2*j]   = tobf(p[16]); b1[2*j+1] = tobf(p[N + 16]);
    }
    c0 = __builtin_amdgcn_wmma_f32_16x16x32_bf16(false, a, false, b0, (short)0, c0, false, false);
    c1 = __builtin_amdgcn_wmma_f32_16x16x32_bf16(false, a, false, b1, (short)0, c1, false, false);
  }

#pragma unroll
  for (int r = 0; r < 8; ++r) {
    int mm = r + (lhalf << 3);
    size_t i0 = (size_t)(row0 + mm) * N + col0 + lrow;
    size_t i1 = i0 + 16;
    float o0 = c0[r], o1 = c1[r];
    if (bias) { o0 += bias[col0 + lrow]; o1 += bias[col0 + 16 + lrow]; }
    if (relu) { o0 = fmaxf(o0, 0.f);     o1 = fmaxf(o1, 0.f); }
    if (Resid) { o0 += Resid[i0];        o1 += Resid[i1]; }
    C[i0] = o0; C[i1] = o1;
  }
}

// ---------------------------------------------------------------------------
// rel_lin[l,r,:] = rel_emb[r,:] @ gine_ew[l] + gine_eb[l]   (tiny: 3*32*256)
// ---------------------------------------------------------------------------
__global__ void rel_lin_kernel(const float* __restrict__ rel_emb,
                               const float* __restrict__ ew,
                               const float* __restrict__ eb,
                               float* __restrict__ out)
{
  int t = blockIdx.x * blockDim.x + threadIdx.x;
  if (t >= Ll * Rr * Hh) return;
  int l = t / (Rr * Hh), rem = t % (Rr * Hh);
  int r = rem / Hh, n = rem % Hh;
  const float* re = rel_emb + (size_t)r * Hh;
  const float* w  = ew + (size_t)l * Hh * Hh;
  float acc = eb[l * Hh + n];
  for (int k = 0; k < Hh; ++k) acc += re[k] * w[(size_t)k * Hh + n];
  out[t] = acc;
}

// ---------------------------------------------------------------------------
// aggr[dst] += relu(x[src] + rel_lin[et])   (64 threads/edge, float4 each)
// ---------------------------------------------------------------------------
__global__ void edge_aggr_kernel(const float* __restrict__ X,
                                 const float* __restrict__ rel_lin,
                                 const int* __restrict__ src,
                                 const int* __restrict__ dst,
                                 const int* __restrict__ et,
                                 float* __restrict__ AGG)
{
  int t = blockIdx.x * blockDim.x + threadIdx.x;
  if (t >= Ee * 64) return;
  int e = t >> 6, f = (t & 63) << 2;
  int s = src[e], d = dst[e], r = et[e];
  float4 xv = *(const float4*)(X + (size_t)s * Hh + f);
  float4 rv = *(const float4*)(rel_lin + (size_t)r * Hh + f);
  float* o = AGG + (size_t)d * Hh + f;
  atomicAdd(o + 0, fmaxf(xv.x + rv.x, 0.f));
  atomicAdd(o + 1, fmaxf(xv.y + rv.y, 0.f));
  atomicAdd(o + 2, fmaxf(xv.z + rv.z, 0.f));
  atomicAdd(o + 3, fmaxf(xv.w + rv.w, 0.f));
}

__global__ void zero_kernel(float4* __restrict__ p, int n4)
{
  int t = blockIdx.x * blockDim.x + threadIdx.x;
  if (t < n4) p[t] = make_float4(0.f, 0.f, 0.f, 0.f);
}

__global__ void add_emb_kernel(float* __restrict__ X, const int* __restrict__ idx,
                               const float* __restrict__ emb, int cnt)
{
  int t = blockIdx.x * blockDim.x + threadIdx.x;
  if (t >= cnt * 64) return;
  int i = t >> 6, f = (t & 63) << 2;
  float4 e = *(const float4*)(emb + f);
  float* p = X + (size_t)idx[i] * Hh + f;
  p[0] += e.x; p[1] += e.y; p[2] += e.z; p[3] += e.w;
}

__global__ void gather_rows_kernel(const float* __restrict__ X, const int* __restrict__ idx,
                                   float* __restrict__ out, int cnt)
{
  int t = blockIdx.x * blockDim.x + threadIdx.x;
  if (t >= cnt * 64) return;
  int i = t >> 6, f = (t & 63) << 2;
  *(float4*)(out + (size_t)i * Hh + f) = *(const float4*)(X + (size_t)idx[i] * Hh + f);
}

__global__ void scatter_rows_kernel(const float* __restrict__ src_, const int* __restrict__ idx,
                                    float* __restrict__ X, int cnt)
{
  int t = blockIdx.x * blockDim.x + threadIdx.x;
  if (t >= cnt * 64) return;
  int i = t >> 6, f = (t & 63) << 2;
  *(float4*)(X + (size_t)idx[i] * Hh + f) = *(const float4*)(src_ + (size_t)i * Hh + f);
}

// ---------------------------------------------------------------------------
// LayerNorm over rows of 256: out = ln(In (+Res)) * g + b.  One wave per row.
// ---------------------------------------------------------------------------
__global__ __launch_bounds__(256) void layernorm_kernel(
    const float* __restrict__ In, const float* __restrict__ Res,
    const float* __restrict__ g, const float* __restrict__ b,
    float* __restrict__ Out, int rows)
{
  int wave = threadIdx.x >> 5, lane = threadIdx.x & 31;
  int row = blockIdx.x * 8 + wave;
  if (row >= rows) return;
  const float* ip = In + (size_t)row * Hh;
  const float* rp = Res ? Res + (size_t)row * Hh : nullptr;
  float v[8], s = 0.f, s2 = 0.f;
#pragma unroll
  for (int i = 0; i < 8; ++i) {
    float x = ip[lane + 32 * i];
    if (rp) x += rp[lane + 32 * i];
    v[i] = x; s += x; s2 += x * x;
  }
#pragma unroll
  for (int m = 1; m < 32; m <<= 1) { s += __shfl_xor(s, m, 32); s2 += __shfl_xor(s2, m, 32); }
  float mean = s * (1.f / Hh);
  float var  = s2 * (1.f / Hh) - mean * mean;
  float inv  = rsqrtf(var + 1e-5f);
#pragma unroll
  for (int i = 0; i < 8; ++i) {
    int c = lane + 32 * i;
    Out[(size_t)row * Hh + c] = (v[i] - mean) * inv * g[c] + b[c];
  }
}

// ---------------------------------------------------------------------------
// Attention: one block per (chunk, head). 8 waves; each wave owns 4 q-tiles
// of 16 rows and streams keys in 32-wide blocks with online softmax.
//   scores: s = Q(A-frag) x K^T(B-frag) WMMA, K-dim = HD = 32.
//   probs -> LDS (bf16) -> reloaded as A-frag for o += P x V WMMA.
// ---------------------------------------------------------------------------
__global__ __launch_bounds__(256) void attn_kernel(const float* __restrict__ QKV,
                                                   float* __restrict__ ATT,
                                                   float scale)
{
  __shared__ bf16 ptile[8][16 * 32];
  const int lane  = threadIdx.x & 31;
  const int wave  = threadIdx.x >> 5;
  const int lrow  = lane & 15;
  const int lhalf = lane >> 4;
  const int chunk = blockIdx.x >> 3;
  const int head  = blockIdx.x & 7;
  const int base  = chunk * BCH;
  const int qcol = head * HD, kcol = Hh + head * HD, vcol = 2 * Hh + head * HD;

  for (int qt = wave; qt < BCH / 16; qt += 8) {
    // Q fragment (A layout): m = lrow, k = head-dim index (two contiguous runs)
    v16bf qa;
    load_a_frag(QKV + (size_t)(base + qt * 16 + lrow) * 768 + qcol + (lhalf << 3), qa);

    float mprev[8], lsum[8];
    v8f acc0 = {}, acc1 = {};
#pragma unroll
    for (int r = 0; r < 8; ++r) { mprev[r] = -3.0e38f; lsum[r] = 0.f; }

    for (int kt = 0; kt < BCH / 32; ++kt) {
      // K^T fragments (B layout): per lane head-dim runs {0..7},{16..23} after lhalf*8.
      v16bf kb0, kb1;
      load_b_frag_contig(QKV + (size_t)(base + kt * 32 + lrow)      * 768 + kcol + (lhalf << 3), kb0);
      load_b_frag_contig(QKV + (size_t)(base + kt * 32 + 16 + lrow) * 768 + kcol + (lhalf << 3), kb1);
      v8f s0 = {}, s1 = {};
      s0 = __builtin_amdgcn_wmma_f32_16x16x32_bf16(false, qa, false, kb0, (short)0, s0, false, false);
      s1 = __builtin_amdgcn_wmma_f32_16x16x32_bf16(false, qa, false, kb1, (short)0, s1, false, false);

      // online softmax over the 32 keys of this block
#pragma unroll
      for (int r = 0; r < 8; ++r) {
        float v0 = s0[r] * scale, v1 = s1[r] * scale;
        float mx = fmaxf(v0, v1);
#pragma unroll
        for (int m = 1; m < 16; m <<= 1) mx = fmaxf(mx, __shfl_xor(mx, m, 32));
        float mnew = fmaxf(mprev[r], mx);
        float ef = __expf(mprev[r] - mnew);
        float p0 = __expf(v0 - mnew), p1 = __expf(v1 - mnew);
        float rs = p0 + p1;
#pragma unroll
        for (int m = 1; m < 16; m <<= 1) rs += __shfl_xor(rs, m, 32);
        lsum[r] = lsum[r] * ef + rs;
        mprev[r] = mnew;
        acc0[r] *= ef; acc1[r] *= ef;
        int mm = r + (lhalf << 3);
        ptile[wave][mm * 32 + lrow]      = tobf(p0);
        ptile[wave][mm * 32 + 16 + lrow] = tobf(p1);
      }
      asm volatile("s_wait_dscnt 0" ::: "memory");

      // P as A-frag (m = q row, k = key), V as B-frags (k = key, n = head dim)
      v16bf pa, vb0, vb1;
#pragma unroll
      for (int i = 0; i < 16; ++i) {
        int ka = a_kidx(i, lhalf);
        pa[i] = ptile[wave][lrow * 32 + ka];
        int kb = b_kidx(i, lhalf);
        const float* vrow = QKV + (size_t)(base + kt * 32 + kb) * 768 + vcol;
        vb0[i] = tobf(vrow[lrow]);
        vb1[i] = tobf(vrow[16 + lrow]);
      }
      acc0 = __builtin_amdgcn_wmma_f32_16x16x32_bf16(false, pa, false, vb0, (short)0, acc0, false, false);
      acc1 = __builtin_amdgcn_wmma_f32_16x16x32_bf16(false, pa, false, vb1, (short)0, acc1, false, false);
    }

#pragma unroll
    for (int r = 0; r < 8; ++r) {
      int mm = r + (lhalf << 3);
      float inv = 1.f / lsum[r];
      size_t row = (size_t)(base + qt * 16 + mm);
      ATT[row * Hh + head * HD + lrow]      = acc0[r] * inv;
      ATT[row * Hh + head * HD + 16 + lrow] = acc1[r] * inv;
    }
  }
}

// ---------------------------------------------------------------------------
// out[N,12] = x @ out_w + out_b  (narrow N: plain VALU; weights pulled into
// LDS with gfx1250 async global->LDS copies)
// ---------------------------------------------------------------------------
__global__ __launch_bounds__(256) void out_proj_kernel(
    const float* __restrict__ X, const float* __restrict__ W,
    const float* __restrict__ bias, float* __restrict__ Out, int rows)
{
  __shared__ float wsm[Hh * 12];   // 12288 bytes
  {
    // 256 threads x 3 x 16B = 12 KB, via async global->LDS (ASYNCcnt).
    unsigned ldsbase = (unsigned)(uintptr_t)&wsm[0];
#pragma unroll
    for (int j = 0; j < 3; ++j) {
      unsigned off = threadIdx.x * 16u + (unsigned)j * 4096u;
      unsigned lds = ldsbase + off;
      const float* gp = W + (off >> 2);
      asm volatile("global_load_async_to_lds_b128 %0, %1, off"
                   :: "v"(lds), "v"(gp) : "memory");
    }
    asm volatile("s_wait_asynccnt 0" ::: "memory");
  }
  __syncthreads();

  int row = blockIdx.x * 256 + threadIdx.x;
  if (row >= rows) return;
  float acc[12];
#pragma unroll
  for (int j = 0; j < 12; ++j) acc[j] = bias[j];
  const float* xr = X + (size_t)row * Hh;
  for (int k = 0; k < Hh; ++k) {
    float xv = xr[k];
#pragma unroll
    for (int j = 0; j < 12; ++j) acc[j] += xv * wsm[k * 12 + j];
  }
#pragma unroll
  for (int j = 0; j < 12; ++j) Out[(size_t)row * 12 + j] = acc[j];
}

// ---------------------------------------------------------------------------
// Host-side orchestration
// ---------------------------------------------------------------------------
static inline void launch_gemm(const float* A, const float* A2, const float* B,
                               const float* bias, const float* R, float* C,
                               int M, int K, int N, int relu, hipStream_t s)
{
  dim3 grid(N / 64, (M + 63) / 64);
  if (A2) gemm_wmma_kernel<true><<<grid, 256, 0, s>>>(A, A2, B, bias, R, C, M, K, N, relu);
  else    gemm_wmma_kernel<false><<<grid, 256, 0, s>>>(A, nullptr, B, bias, R, C, M, K, N, relu);
}

extern "C" void kernel_launch(void* const* d_in, const int* in_sizes, int n_in,
                              void* d_out, int out_size, void* d_ws, size_t ws_size,
                              hipStream_t stream)
{
  (void)n_in; (void)out_size; (void)ws_size;
  const float* x_in   = (const float*)d_in[0];
  const float* np_w   = (const float*)d_in[1];
  const float* np_b   = (const float*)d_in[2];
  const float* g_emb  = (const float*)d_in[3];
  const float* p_emb  = (const float*)d_in[4];
  const float* relE   = (const float*)d_in[5];
  const float* gw1    = (const float*)d_in[6];
  const float* gb1    = (const float*)d_in[7];
  const float* gw2    = (const float*)d_in[8];
  const float* gb2    = (const float*)d_in[9];
  const float* gew    = (const float*)d_in[10];
  const float* geb    = (const float*)d_in[11];
  const float* n1g    = (const float*)d_in[12];
  const float* n1b    = (const float*)d_in[13];
  const float* wqkv   = (const float*)d_in[14];
  const float* bqkv   = (const float*)d_in[15];
  const float* wo     = (const float*)d_in[16];
  const float* bo     = (const float*)d_in[17];
  const float* n2g    = (const float*)d_in[18];
  const float* n2b    = (const float*)d_in[19];
  const float* fw1    = (const float*)d_in[20];
  const float* fb1    = (const float*)d_in[21];
  const float* fw2    = (const float*)d_in[22];
  const float* fb2    = (const float*)d_in[23];
  const float* outw   = (const float*)d_in[24];
  const float* outb   = (const float*)d_in[25];
  const int*   ei     = (const int*)d_in[26];
  const int*   et     = (const int*)d_in[27];
  const int*   chem   = (const int*)d_in[28];
  const int*   geneI  = (const int*)d_in[29];
  const int*   pathI  = (const int*)d_in[30];
  const int gcnt = in_sizes[29];
  const int pcnt = in_sizes[30];
  const int* src = ei;          // ei[0]
  const int* dst = ei + Ee;     // ei[1]

  // workspace carve-up (floats)
  float* ws = (float*)d_ws;
  const size_t NH = (size_t)Nn * Hh;
  float* X    = ws;
  float* AGG  = X    + NH;
  float* TMP  = AGG  + NH;
  float* QKV  = TMP  + NH;
  float* ATT  = QKV  + (size_t)NCc * 3 * Hh;
  float* XC   = ATT  + (size_t)NCc * Hh;
  float* FFNT = XC   + (size_t)NCc * Hh;
  float* RLIN = FFNT + (size_t)NCc * 2 * Hh;

  const float scale = 1.f / sqrtf((float)HD);

  // x = x @ node_proj_w + b
  launch_gemm(x_in, nullptr, np_w, np_b, nullptr, X, Nn, INF_, Hh, 0, stream);
  // x[gene_idx] += gene_emb ; x[path_idx] += path_emb
  add_emb_kernel<<<(gcnt * 64 + 255) / 256, 256, 0, stream>>>(X, geneI, g_emb, gcnt);
  add_emb_kernel<<<(pcnt * 64 + 255) / 256, 256, 0, stream>>>(X, pathI, p_emb, pcnt);
  // rel_lin[l] = rel_emb @ gine_ew[l] + gine_eb[l]
  rel_lin_kernel<<<(Ll * Rr * Hh + 255) / 256, 256, 0, stream>>>(relE, gew, geb, RLIN);

  for (int l = 0; l < Ll; ++l) {
    const size_t lHH = (size_t)l * Hh * Hh;
    // GINE aggregation
    zero_kernel<<<((int)(NH / 4) + 255) / 256, 256, 0, stream>>>((float4*)AGG, (int)(NH / 4));
    edge_aggr_kernel<<<(Ee * 64 + 255) / 256, 256, 0, stream>>>(
        X, RLIN + (size_t)l * Rr * Hh, src, dst, et, AGG);
    // h = relu((x+aggr)@W1+b1) @ W2 + b2 ; x = ln(x + h)
    launch_gemm(X, AGG, gw1 + lHH, gb1 + l * Hh, nullptr, TMP, Nn, Hh, Hh, 1, stream);
    launch_gemm(TMP, nullptr, gw2 + lHH, gb2 + l * Hh, nullptr, AGG, Nn, Hh, Hh, 0, stream);
    layernorm_kernel<<<Nn / 8, 256, 0, stream>>>(AGG, X, n1g + l * Hh, n1b + l * Hh, X, Nn);
    // chem attention
    gather_rows_kernel<<<(NCc * 64 + 255) / 256, 256, 0, stream>>>(X, chem, XC, NCc);
    launch_gemm(XC, nullptr, wqkv + (size_t)l * Hh * 3 * Hh, bqkv + l * 3 * Hh,
                nullptr, QKV, NCc, Hh, 3 * Hh, 0, stream);
    attn_kernel<<<NCHUNK * HEADS, 256, 0, stream>>>(QKV, ATT, scale);
    launch_gemm(ATT, nullptr, wo + lHH, bo + l * Hh, XC, XC, NCc, Hh, Hh, 0, stream);
    // FFN: xc += relu(ln(xc)@W1+b1)@W2+b2
    layernorm_kernel<<<NCc / 8, 256, 0, stream>>>(XC, nullptr, n2g + l * Hh, n2b + l * Hh, ATT, NCc);
    launch_gemm(ATT, nullptr, fw1 + (size_t)l * Hh * 2 * Hh, fb1 + l * 2 * Hh,
                nullptr, FFNT, NCc, Hh, 2 * Hh, 1, stream);
    launch_gemm(FFNT, nullptr, fw2 + (size_t)l * 2 * Hh * Hh, fb2 + l * Hh,
                XC, XC, NCc, 2 * Hh, Hh, 0, stream);
    scatter_rows_kernel<<<(NCc * 64 + 255) / 256, 256, 0, stream>>>(XC, chem, X, NCc);
  }

  // final projection to 12 outputs
  out_proj_kernel<<<(Nn + 255) / 256, 256, 0, stream>>>(X, outw, outb, (float*)d_out, Nn);
}
